// OptimalTransportCenterlines_41755672052331
// MI455X (gfx1250) — compile-verified, matrix operand order checked by reading, not run on hardware
//
#include <hip/hip_runtime.h>
#include <hip/hip_bf16.h>
#include <math.h>

typedef __attribute__((ext_vector_type(16))) _Float16 v16h;
typedef __attribute__((ext_vector_type(8)))  float    v8f;

#define BB 8
#define NN 1500
#define NP 1501
#define DD 256
#define K1 512
#define MROWS (BB * NN)   /* 12000 */
#define SINK_ITERS 50

#define ST_M 47           /* ceil(1500/32) 32-row score tiles  */
#define ST_N 24           /* ceil(1500/64) 64-col score tiles  */

// ---------------------------------------------------------------------------
// WMMA fragment loaders: 16-bit A/B fragment, source row-major, K contiguous.
// Per ISA 16-bit 16x32 layout:
//   lane L: row/col = L&15 ; halves h: k = (h>=8)*16 + ((L>>4)&1)*8 + (h&7)
// -> two contiguous 8-half (16B) chunks per lane.
__device__ __forceinline__ v16h ot_ldfrag_u(const _Float16* base, int stride,
                                            int idx0, int k0, int lane) {
  const int grp = (lane >> 4) & 1;
  const int r = idx0 + (lane & 15);
  const _Float16* p = base + (size_t)r * stride + k0 + grp * 8;
  union { uint4 q[2]; v16h h; } cvt;
  cvt.q[0] = *reinterpret_cast<const uint4*>(p);
  cvt.q[1] = *reinterpret_cast<const uint4*>(p + 16);
  return cvt.h;
}

__device__ __forceinline__ v16h ot_ldfrag_g(const _Float16* base, int stride,
                                            int idx0, int k0, int lane, int idxMax) {
  const int grp = (lane >> 4) & 1;
  const int r = idx0 + (lane & 15);
  v16h f;
  if (r < idxMax) {
    const _Float16* p = base + (size_t)r * stride + k0 + grp * 8;
    union { uint4 q[2]; v16h h; } cvt;
    cvt.q[0] = *reinterpret_cast<const uint4*>(p);
    cvt.q[1] = *reinterpret_cast<const uint4*>(p + 16);
    f = cvt.h;
  } else {
#pragma unroll
    for (int i = 0; i < 16; ++i) f[i] = (_Float16)0.0f;
  }
  return f;
}

__device__ __forceinline__ v8f ot_wmma(v16h a, v16h b, v8f c) {
  return __builtin_amdgcn_wmma_f32_16x16x32_f16(false, a, false, b, (short)0, c,
                                                false, false);
}

// ---------------------------------------------------------------------------
// Prep kernels
__global__ void ot_pack_wt(const float* __restrict__ W, _Float16* __restrict__ Wt,
                           int K, int Nout) {
  int t = blockIdx.x * 256 + threadIdx.x;
  if (t >= K * Nout) return;
  int o = t / K, k = t - o * K;
  Wt[t] = (_Float16)W[(size_t)k * Nout + o];   // W is [K][Nout] -> Wt is [Nout][K]
}

__global__ void ot_build_xcat(const float* __restrict__ q, const float* __restrict__ p,
                              _Float16* __restrict__ X) {
  size_t t = (size_t)blockIdx.x * 256 + threadIdx.x;
  if (t >= (size_t)MROWS * K1) return;
  size_t row = t >> 9;
  int c = (int)(t & 511);
  float val = (c < DD) ? q[row * DD + c] : p[row * DD + (c - DD)];
  X[t] = (_Float16)val;
}

// ---------------------------------------------------------------------------
// MLP layer: out[M x 256] = act(X[M x K] @ Wt^T + bias).
// REQUIRES M % 32 == 0 (true for all uses: M = 12000).
// Each wave computes a 32x64 tile: 8 WMMAs per K-step, A reused 4x, B 2x.
// No bounds checks anywhere -> clean clause of 12 b128 loads + 8 WMMAs.
__global__ void ot_mlp_gemm(const _Float16* __restrict__ X, const _Float16* __restrict__ Wt,
                            const float* __restrict__ bias, _Float16* __restrict__ out,
                            int M, int K, int relu) {
  int wid = (int)((blockIdx.x * blockDim.x + threadIdx.x) >> 5);
  int lane = threadIdx.x & 31;
  int tilesM = M >> 5;
  int tM = wid >> 2;      // 4 N-groups of 64 cols (256/64)
  int tN = wid & 3;
  if (tM >= tilesM) return;
  const int m0 = tM * 32, n0 = tN * 64;

  v8f acc[2][4];
#pragma unroll
  for (int mi = 0; mi < 2; ++mi)
#pragma unroll
    for (int ni = 0; ni < 4; ++ni) acc[mi][ni] = (v8f){};

#pragma unroll 1
  for (int k0 = 0; k0 < K; k0 += 32) {
    v16h a0 = ot_ldfrag_u(X, K, m0,      k0, lane);
    v16h a1 = ot_ldfrag_u(X, K, m0 + 16, k0, lane);
    v16h b0 = ot_ldfrag_u(Wt, K, n0,      k0, lane);
    v16h b1 = ot_ldfrag_u(Wt, K, n0 + 16, k0, lane);
    v16h b2 = ot_ldfrag_u(Wt, K, n0 + 32, k0, lane);
    v16h b3 = ot_ldfrag_u(Wt, K, n0 + 48, k0, lane);
    acc[0][0] = ot_wmma(a0, b0, acc[0][0]);
    acc[0][1] = ot_wmma(a0, b1, acc[0][1]);
    acc[0][2] = ot_wmma(a0, b2, acc[0][2]);
    acc[0][3] = ot_wmma(a0, b3, acc[0][3]);
    acc[1][0] = ot_wmma(a1, b0, acc[1][0]);
    acc[1][1] = ot_wmma(a1, b1, acc[1][1]);
    acc[1][2] = ot_wmma(a1, b2, acc[1][2]);
    acc[1][3] = ot_wmma(a1, b3, acc[1][3]);
  }

#pragma unroll
  for (int ni = 0; ni < 4; ++ni) {
    int n = n0 + ni * 16 + (lane & 15);
    float bv = bias[n];
#pragma unroll
    for (int mi = 0; mi < 2; ++mi) {
#pragma unroll
      for (int r = 0; r < 8; ++r) {
        int m = m0 + mi * 16 + r + ((lane >> 4) << 3);
        float x = acc[mi][ni][r] + bv;
        if (relu) x = fmaxf(x, 0.0f);
        out[(size_t)m * DD + n] = (_Float16)x;
      }
    }
  }
}

// ---------------------------------------------------------------------------
// scores[b,i,j] = (F1[b,i,:] . F2[b,j,:]) / 16, written into C[b, NP, NP].
// 32x64 tile per wave.  Interior tiles (46/47 x 23/24 of all tiles) take a
// fully unguarded path; only edge tiles pay for bounds handling.
template <bool GUARD>
__device__ __forceinline__ void ot_score_accum(const _Float16* A, const _Float16* Bm,
                                               int m0, int n0, int lane,
                                               v8f acc[2][4]) {
#pragma unroll 1
  for (int k0 = 0; k0 < DD; k0 += 32) {
    v16h a0, a1, b0, b1, b2, b3;
    if (GUARD) {
      a0 = ot_ldfrag_g(A, DD, m0,      k0, lane, NN);
      a1 = ot_ldfrag_g(A, DD, m0 + 16, k0, lane, NN);
      b0 = ot_ldfrag_g(Bm, DD, n0,      k0, lane, NN);
      b1 = ot_ldfrag_g(Bm, DD, n0 + 16, k0, lane, NN);
      b2 = ot_ldfrag_g(Bm, DD, n0 + 32, k0, lane, NN);
      b3 = ot_ldfrag_g(Bm, DD, n0 + 48, k0, lane, NN);
    } else {
      a0 = ot_ldfrag_u(A, DD, m0,      k0, lane);
      a1 = ot_ldfrag_u(A, DD, m0 + 16, k0, lane);
      b0 = ot_ldfrag_u(Bm, DD, n0,      k0, lane);
      b1 = ot_ldfrag_u(Bm, DD, n0 + 16, k0, lane);
      b2 = ot_ldfrag_u(Bm, DD, n0 + 32, k0, lane);
      b3 = ot_ldfrag_u(Bm, DD, n0 + 48, k0, lane);
    }
    acc[0][0] = ot_wmma(a0, b0, acc[0][0]);
    acc[0][1] = ot_wmma(a0, b1, acc[0][1]);
    acc[0][2] = ot_wmma(a0, b2, acc[0][2]);
    acc[0][3] = ot_wmma(a0, b3, acc[0][3]);
    acc[1][0] = ot_wmma(a1, b0, acc[1][0]);
    acc[1][1] = ot_wmma(a1, b1, acc[1][1]);
    acc[1][2] = ot_wmma(a1, b2, acc[1][2]);
    acc[1][3] = ot_wmma(a1, b3, acc[1][3]);
  }
}

__global__ void ot_scores(const _Float16* __restrict__ F1, const _Float16* __restrict__ F2,
                          float* __restrict__ C) {
  int wid = (int)((blockIdx.x * blockDim.x + threadIdx.x) >> 5);
  int lane = threadIdx.x & 31;
  int b = wid / (ST_M * ST_N);
  if (b >= BB) return;
  int t = wid - b * ST_M * ST_N;
  int tM = t / ST_N, tN = t - tM * ST_N;
  const int m0 = tM * 32, n0 = tN * 64;
  const _Float16* A  = F1 + (size_t)b * NN * DD;
  const _Float16* Bm = F2 + (size_t)b * NN * DD;

  v8f acc[2][4];
#pragma unroll
  for (int mi = 0; mi < 2; ++mi)
#pragma unroll
    for (int ni = 0; ni < 4; ++ni) acc[mi][ni] = (v8f){};

  const bool interior = (m0 + 32 <= NN) && (n0 + 64 <= NN);  // wave-uniform
  float* Cb = C + (size_t)b * NP * NP;

  if (interior) {
    ot_score_accum<false>(A, Bm, m0, n0, lane, acc);
#pragma unroll
    for (int ni = 0; ni < 4; ++ni) {
      int n = n0 + ni * 16 + (lane & 15);
#pragma unroll
      for (int mi = 0; mi < 2; ++mi) {
#pragma unroll
        for (int r = 0; r < 8; ++r) {
          int m = m0 + mi * 16 + r + ((lane >> 4) << 3);
          Cb[(size_t)m * NP + n] = acc[mi][ni][r] * 0.0625f;
        }
      }
    }
  } else {
    ot_score_accum<true>(A, Bm, m0, n0, lane, acc);
#pragma unroll
    for (int ni = 0; ni < 4; ++ni) {
      int n = n0 + ni * 16 + (lane & 15);
#pragma unroll
      for (int mi = 0; mi < 2; ++mi) {
#pragma unroll
        for (int r = 0; r < 8; ++r) {
          int m = m0 + mi * 16 + r + ((lane >> 4) << 3);
          if (m < NN && n < NN) Cb[(size_t)m * NP + n] = acc[mi][ni][r] * 0.0625f;
        }
      }
    }
  }
}

// bin row/col = alpha, diagonal = -inf
__global__ void ot_fill_bins(float* __restrict__ C, const float* __restrict__ alpha_p) {
  int t = blockIdx.x * 256 + threadIdx.x;
  if (t >= BB * NP) return;
  int b = t / NP, x = t - b * NP;
  float alpha = *alpha_p;
  float* Cb = C + (size_t)b * NP * NP;
  Cb[(size_t)NN * NP + x] = alpha;
  if (x < NN) {
    Cb[(size_t)x * NP + NN] = alpha;
    Cb[(size_t)x * NP + x] = -__builtin_inff();
  }
}

__global__ void ot_zero_uv(float* __restrict__ u, float* __restrict__ v) {
  int t = blockIdx.x * 256 + threadIdx.x;
  if (t >= BB * NP) return;
  u[t] = 0.0f;
  v[t] = 0.0f;
}

// ---------------------------------------------------------------------------
// u[b,i] = log_mu[i] - logsumexp_j(C[b,i,j] + v[b,j]);  one block per (b,i)
__global__ void ot_row_pass(const float* __restrict__ C, const float* __restrict__ v,
                            float* __restrict__ u) {
  int blk = blockIdx.x;
  int b = blk / NP, i = blk - b * NP;
  const float* row = C + ((size_t)b * NP + i) * NP;
  const float* vb = v + b * NP;
  int t = threadIdx.x;
  float x[6];
#pragma unroll
  for (int k = 0; k < 6; ++k) {
    int j = t + k * 256;
    x[k] = (j < NP) ? row[j] + vb[j] : -__builtin_inff();
  }
  float M = x[0];
#pragma unroll
  for (int k = 1; k < 6; ++k) M = fmaxf(M, x[k]);
  // Each thread holds >=5 in-range values, at most one -inf per row -> M finite.
  float S = 0.0f;
#pragma unroll
  for (int k = 0; k < 6; ++k) S += __expf(x[k] - M);
  __shared__ float sM[256], sS[256];
  sM[t] = M; sS[t] = S;
  __syncthreads();
  for (int s = 128; s > 0; s >>= 1) {
    if (t < s) {
      float Ma = sM[t], Mb = sM[t + s];
      float Mx = fmaxf(Ma, Mb);
      sS[t] = sS[t] * __expf(Ma - Mx) + sS[t + s] * __expf(Mb - Mx);
      sM[t] = Mx;
    }
    __syncthreads();
  }
  if (t == 0) {
    float norm = -__logf((float)(NN + NN));
    float lmu = (i < NN) ? norm : __logf((float)NN) + norm;
    u[b * NP + i] = lmu - (sM[0] + __logf(sS[0]));
  }
}

// v[b,j] = log_nu[j] - logsumexp_i(C[b,i,j] + u[b,i])
// Block = 32 columns x 8 row-partitions (coalesced along j), online LSE.
__global__ void ot_col_pass(const float* __restrict__ C, const float* __restrict__ u,
                            float* __restrict__ v) {
  int cx = threadIdx.x & 31, ry = threadIdx.x >> 5;
  int j = blockIdx.x * 32 + cx;
  int b = blockIdx.y;
  const float* Cb = C + (size_t)b * NP * NP;
  const float* ub = u + b * NP;
  float M = -__builtin_inff(), S = 0.0f;
  if (j < NP) {
    for (int i = ry; i < NP; i += 8) {
      float x = Cb[(size_t)i * NP + j] + ub[i];
      if (x == -__builtin_inff()) continue;   // diagonal entry
      if (x > M) { S = S * __expf(M - x) + 1.0f; M = x; }
      else       { S += __expf(x - M); }
    }
  }
  __shared__ float sM[256], sS[256];
  int t = threadIdx.x;
  sM[t] = M; sS[t] = S;
  __syncthreads();
  for (int off = 128; off >= 32; off >>= 1) {
    if (t < off) {
      float Ma = sM[t], Mb = sM[t + off];
      float Mx = fmaxf(Ma, Mb);
      float Sx = 0.0f;
      if (Mx > -__builtin_inff())
        Sx = sS[t] * __expf(Ma - Mx) + sS[t + off] * __expf(Mb - Mx);
      sM[t] = Mx; sS[t] = Sx;
    }
    __syncthreads();
  }
  if (ry == 0 && j < NP) {
    float norm = -__logf((float)(NN + NN));
    float lnu = (j < NN) ? norm : __logf((float)NN) + norm;
    v[b * NP + j] = lnu - (sM[cx] + __logf(sS[cx]));
  }
}

// out = exp(C + u + v - norm), in place on d_out
__global__ void ot_finalize(float* __restrict__ C, const float* __restrict__ u,
                            const float* __restrict__ v) {
  size_t t = (size_t)blockIdx.x * 256 + threadIdx.x;
  const size_t per = (size_t)NP * NP;
  if (t >= (size_t)BB * per) return;
  int b = (int)(t / per);
  size_t r = t - (size_t)b * per;
  int i = (int)(r / NP);
  int j = (int)(r - (size_t)i * NP);
  float norm = -__logf((float)(NN + NN));
  C[t] = __expf(C[t] + u[b * NP + i] + v[b * NP + j] - norm);
}

// ---------------------------------------------------------------------------
extern "C" void kernel_launch(void* const* d_in, const int* in_sizes, int n_in,
                              void* d_out, int out_size, void* d_ws, size_t ws_size,
                              hipStream_t stream) {
  (void)in_sizes; (void)n_in; (void)out_size; (void)ws_size;
  const float* quer      = (const float*)d_in[0];
  const float* pos_start = (const float*)d_in[1];
  const float* pos_end   = (const float*)d_in[2];
  const float* W1 = (const float*)d_in[3];
  const float* b1 = (const float*)d_in[4];
  const float* W2 = (const float*)d_in[5];
  const float* b2 = (const float*)d_in[6];
  const float* W3 = (const float*)d_in[7];
  const float* b3 = (const float*)d_in[8];
  const float* bin = (const float*)d_in[9];
  float* C = (float*)d_out;   // couplings live in d_out, finalized in place

  char* p = (char*)d_ws;
  auto carve = [&](size_t bytes) -> char* {
    char* r = p; p += (bytes + 255) & ~(size_t)255; return r;
  };
  _Float16* Wt1 = (_Float16*)carve((size_t)K1 * DD * 2);
  _Float16* Wt2 = (_Float16*)carve((size_t)DD * DD * 2);
  _Float16* Wt3 = (_Float16*)carve((size_t)DD * DD * 2);
  _Float16* Xc  = (_Float16*)carve((size_t)MROWS * K1 * 2);
  _Float16* H1  = (_Float16*)carve((size_t)MROWS * DD * 2);
  _Float16* H2  = (_Float16*)carve((size_t)MROWS * DD * 2);
  _Float16* F1  = (_Float16*)carve((size_t)MROWS * DD * 2);
  _Float16* F2  = (_Float16*)carve((size_t)MROWS * DD * 2);
  float* u = (float*)carve((size_t)BB * NP * 4);
  float* v = (float*)carve((size_t)BB * NP * 4);

  ot_pack_wt<<<(K1 * DD + 255) / 256, 256, 0, stream>>>(W1, Wt1, K1, DD);
  ot_pack_wt<<<(DD * DD + 255) / 256, 256, 0, stream>>>(W2, Wt2, DD, DD);
  ot_pack_wt<<<(DD * DD + 255) / 256, 256, 0, stream>>>(W3, Wt3, DD, DD);

  // MLP GEMM: (M/32) tiles * 4 N-groups waves, 8 waves/block
  const unsigned mlp_waves  = (MROWS / 32) * 4;                  // 1500
  const unsigned mlp_blocks = (mlp_waves * 32 + 255) / 256;      // 188
  const size_t xcat_total = (size_t)MROWS * K1;
  const unsigned xcat_blocks = (unsigned)((xcat_total + 255) / 256);

  // f1 = MLP([quer, pos_end])
  ot_build_xcat<<<xcat_blocks, 256, 0, stream>>>(quer, pos_end, Xc);
  ot_mlp_gemm<<<mlp_blocks, 256, 0, stream>>>(Xc, Wt1, b1, H1, MROWS, K1, 1);
  ot_mlp_gemm<<<mlp_blocks, 256, 0, stream>>>(H1, Wt2, b2, H2, MROWS, DD, 1);
  ot_mlp_gemm<<<mlp_blocks, 256, 0, stream>>>(H2, Wt3, b3, F1, MROWS, DD, 0);
  // f2 = MLP([quer, pos_start])
  ot_build_xcat<<<xcat_blocks, 256, 0, stream>>>(quer, pos_start, Xc);
  ot_mlp_gemm<<<mlp_blocks, 256, 0, stream>>>(Xc, Wt1, b1, H1, MROWS, K1, 1);
  ot_mlp_gemm<<<mlp_blocks, 256, 0, stream>>>(H1, Wt2, b2, H2, MROWS, DD, 1);
  ot_mlp_gemm<<<mlp_blocks, 256, 0, stream>>>(H2, Wt3, b3, F2, MROWS, DD, 0);

  // scores: 8 batches * 47 * 24 waves = 9024 waves -> 1128 blocks
  const unsigned sc_waves  = BB * ST_M * ST_N;
  const unsigned sc_blocks = (sc_waves * 32 + 255) / 256;
  ot_scores<<<sc_blocks, 256, 0, stream>>>(F1, F2, C);
  ot_fill_bins<<<(BB * NP + 255) / 256, 256, 0, stream>>>(C, bin);
  ot_zero_uv<<<(BB * NP + 255) / 256, 256, 0, stream>>>(u, v);

  dim3 colGrid((NP + 31) / 32, BB);
  for (int it = 0; it < SINK_ITERS; ++it) {
    ot_row_pass<<<BB * NP, 256, 0, stream>>>(C, v, u);
    ot_col_pass<<<colGrid, 256, 0, stream>>>(C, u, v);
  }
  const size_t total = (size_t)BB * NP * NP;
  ot_finalize<<<(unsigned)((total + 255) / 256), 256, 0, stream>>>(C, u, v);
}
